// Attention_6579889897835
// MI455X (gfx1250) — compile-verified, matrix operand order checked by reading, not run on hardware
//
#include <hip/hip_runtime.h>
#include <hip/hip_bf16.h>

#define USE_ASYNC 1

typedef __bf16 bf16_t;
typedef __attribute__((ext_vector_type(2)))  __bf16 v2bf;
typedef __attribute__((ext_vector_type(4)))  __bf16 v4bf;
typedef __attribute__((ext_vector_type(8)))  __bf16 v8bf;
typedef __attribute__((ext_vector_type(16))) __bf16 v16bf;
typedef __attribute__((ext_vector_type(8)))  float  v8f;

#define ZERO8F {0.f,0.f,0.f,0.f,0.f,0.f,0.f,0.f}

// ---------------------------------------------------------------------------
// CDNA5 async copy: global -> LDS, 16B per lane, tracked by ASYNCcnt.
// ---------------------------------------------------------------------------
__device__ __forceinline__ void async_load_b128(const bf16_t* gptr, bf16_t* lptr) {
#if USE_ASYNC
    unsigned ldsoff = (unsigned)(size_t)lptr;   // low 32 bits of a shared ptr = LDS offset
    asm volatile("global_load_async_to_lds_b128 %0, %1, off"
                 :: "v"(ldsoff), "v"(gptr) : "memory");
#else
    *(v8bf*)lptr = *(const v8bf*)gptr;
#endif
}
__device__ __forceinline__ void wait_async0() {
#if USE_ASYNC
    asm volatile("s_wait_asynccnt 0x0" ::: "memory");
#endif
}

// 16x32 bf16 WMMA A-fragment (or symmetric B-fragment) from a tile with row
// stride `stride` elements. CDNA5 layout:
//   lanes 0-15 : row = lane,     K = {0..7, 16..23}
//   lanes 16-31: row = lane-16,  K = {8..15, 24..31}
__device__ __forceinline__ v16bf load_frag_b16(const bf16_t* base, int stride) {
    int lane = threadIdx.x & 31;
    const bf16_t* p = base + (lane & 15) * stride + ((lane >> 4) << 3);
    v8bf lo = *(const v8bf*)(p);
    v8bf hi = *(const v8bf*)(p + 16);
    return __builtin_shufflevector(lo, hi, 0,1,2,3,4,5,6,7,8,9,10,11,12,13,14,15);
}

__device__ __forceinline__ v8f wmma_bf16(v16bf a, v16bf b, v8f c) {
    return __builtin_amdgcn_wmma_f32_16x16x32_bf16(false, a, false, b, (short)0, c, false, false);
}

// ---------------------------------------------------------------------------
// One-shot f32 -> bf16 conversion.
// ---------------------------------------------------------------------------
__global__ __launch_bounds__(256)
void cvt_f32_bf16(const float* __restrict__ src, bf16_t* __restrict__ dst, int n4) {
    int i = blockIdx.x * 256 + threadIdx.x;
    if (i < n4) {
        float4 f = ((const float4*)src)[i];
        v4bf o;
        o[0] = (bf16_t)f.x; o[1] = (bf16_t)f.y; o[2] = (bf16_t)f.z; o[3] = (bf16_t)f.w;
        ((v4bf*)dst)[i] = o;
    }
}

// ---------------------------------------------------------------------------
// GEMM: C[M,N](f32) = A[M,K](bf16) @ B[K,N](bf16).
// Block tile 256x128, 256 threads = 8 waves in a 4x2 grid, 64x64 per wave
// (4x4 WMMA accumulators -> 16 WMMAs per k-step per wave).
// Double-buffered LDS; A via global_load_async_to_lds_b128 (4x16B per thread),
// B prefetched to VGPRs and stored transposed as packed v2bf.
// One barrier per K-step. Requires M%256==0, N%128==0, K%32==0.
// ---------------------------------------------------------------------------
__global__ __launch_bounds__(256)
void gemm_bf16_wmma(const bf16_t* __restrict__ A, const bf16_t* __restrict__ B,
                    float* __restrict__ C, int M, int N, int K) {
    __shared__ __align__(16) bf16_t sA [2][256 * 40];   // [row][k]  (80B stride)
    __shared__ __align__(16) bf16_t sBt[2][128 * 40];   // [n][k] transposed

    const int t    = threadIdx.x;
    const int lane = t & 31;
    const int wid  = t >> 5;
    const int wm   = wid >> 1;            // 0..3  (64 rows each)
    const int wn   = wid & 1;             // 0..1  (64 cols each)
    const int bm   = blockIdx.y * 256;
    const int bn   = blockIdx.x * 128;

    // A staging: one 32-elem row per thread, four 16B async copies.
    const int ar = t;
    // B staging: rows (bk,bk+1), 8 cols at bc; packed-pair LDS stores.
    const int bk = (t >> 4) * 2, bc = (t & 15) * 8;

    const bf16_t* gA  = A + (size_t)(bm + ar) * K;
    const bf16_t* gB0 = B + (size_t)bk * N + bn + bc;
    const bf16_t* gB1 = gB0 + N;

    v8f acc[4][4];
    #pragma unroll
    for (int mi = 0; mi < 4; ++mi)
        #pragma unroll
        for (int ni = 0; ni < 4; ++ni) { v8f z = ZERO8F; acc[mi][ni] = z; }

    // ---- prologue: stage k0 = 0 into buffer 0 ----
    #pragma unroll
    for (int u = 0; u < 4; ++u)
        async_load_b128(gA + u * 8, &sA[0][ar * 40 + u * 8]);
    {
        v8bf r0 = *(const v8bf*)gB0;
        v8bf r1 = *(const v8bf*)gB1;
        #pragma unroll
        for (int j = 0; j < 8; ++j) {
            v2bf pk; pk[0] = r0[j]; pk[1] = r1[j];
            *(v2bf*)(&sBt[0][(bc + j) * 40 + bk]) = pk;
        }
    }
    wait_async0();
    __syncthreads();

    int cur = 0;
    for (int k0 = 0; k0 < K; k0 += 32) {
        const int nxt = cur ^ 1;
        const bool more = (k0 + 32) < K;

        v8bf r0, r1;
        if (more) {
            // Issue next-tile copies now; overlap with WMMA work below.
            const bf16_t* gA2 = gA + (k0 + 32);
            #pragma unroll
            for (int u = 0; u < 4; ++u)
                async_load_b128(gA2 + u * 8, &sA[nxt][ar * 40 + u * 8]);
            r0 = *(const v8bf*)(gB0 + (size_t)(k0 + 32) * N);
            r1 = *(const v8bf*)(gB1 + (size_t)(k0 + 32) * N);
        }

        // ---- compute current buffer: 16 WMMAs from 8 fragments ----
        v16bf af[4], bfar[4];
        #pragma unroll
        for (int mi = 0; mi < 4; ++mi)
            af[mi] = load_frag_b16(&sA[cur][(wm * 64 + mi * 16) * 40], 40);
        #pragma unroll
        for (int ni = 0; ni < 4; ++ni)
            bfar[ni] = load_frag_b16(&sBt[cur][(wn * 64 + ni * 16) * 40], 40);
        #pragma unroll
        for (int mi = 0; mi < 4; ++mi)
            #pragma unroll
            for (int ni = 0; ni < 4; ++ni)
                acc[mi][ni] = wmma_bf16(af[mi], bfar[ni], acc[mi][ni]);

        if (more) {
            #pragma unroll
            for (int j = 0; j < 8; ++j) {
                v2bf pk; pk[0] = r0[j]; pk[1] = r1[j];
                *(v2bf*)(&sBt[nxt][(bc + j) * 40 + bk]) = pk;
            }
            wait_async0();
        }
        __syncthreads();
        cur = nxt;
    }

    // Epilogue: lanes 0-15 -> rows i, lanes 16-31 -> rows i+8.
    const int rb = (lane >> 4) * 8;
    const int cn = lane & 15;
    #pragma unroll
    for (int mi = 0; mi < 4; ++mi)
        #pragma unroll
        for (int ni = 0; ni < 4; ++ni) {
            float* cp = C + (size_t)(bm + wm * 64 + mi * 16 + rb) * N
                          + bn + wn * 64 + ni * 16 + cn;
            #pragma unroll
            for (int i = 0; i < 8; ++i) cp[(size_t)i * N] = acc[mi][ni][i];
        }
}

// ---------------------------------------------------------------------------
// RoPE + repack to attention-friendly bf16 layouts.
// ---------------------------------------------------------------------------
__global__ __launch_bounds__(128)
void rope_pack(const float* __restrict__ qf, const float* __restrict__ kf,
               const float* __restrict__ vf,
               bf16_t* __restrict__ Qb, bf16_t* __restrict__ Kb, bf16_t* __restrict__ Vb) {
    const int row = blockIdx.x;
    const int b = row >> 11, s = row & 2047;
    const float NEG_LOG_OVER_32 = -13.122363377404328f / 32.0f;  // -ln(500000)/32
    const float fs = (float)s;

    for (int idx = threadIdx.x; idx < 2048; idx += 128) {
        int h = idx >> 6, d = idx & 63, j = d & 31;
        float ang = fs * __expf((float)j * NEG_LOG_OVER_32);
        float sn, cs; __sincosf(ang, &sn, &cs);
        const float* qr = qf + (size_t)row * 2048;
        float x0 = qr[idx];
        float xr = (d < 32) ? -qr[idx + 32] : qr[idx - 32];
        Qb[(((size_t)(b * 32 + h)) * 2048 + s) * 64 + d] = (bf16_t)(x0 * cs + xr * sn);
    }
    for (int idx = threadIdx.x; idx < 512; idx += 128) {
        int h = idx >> 6, d = idx & 63, j = d & 31;
        float ang = fs * __expf((float)j * NEG_LOG_OVER_32);
        float sn, cs; __sincosf(ang, &sn, &cs);
        const float* kr = kf + (size_t)row * 512;
        float x0 = kr[idx];
        float xr = (d < 32) ? -kr[idx + 32] : kr[idx - 32];
        Kb[(((size_t)(b * 8 + h)) * 2048 + s) * 64 + d] = (bf16_t)(x0 * cs + xr * sn);
    }
    for (int idx = threadIdx.x; idx < 512; idx += 128) {
        int h = idx >> 6, d = idx & 63;
        Vb[(((size_t)(b * 8 + h)) * 2048 + s) * 64 + d] = (bf16_t)vf[(size_t)row * 512 + idx];
    }
}

// ---------------------------------------------------------------------------
// Flash attention: block = (b, head, 128 q-rows), 4 waves x 32 rows
// (two 16-row subtiles per wave -> 16 WMMAs per 32-key step per wave).
// K tile staged async-to-LDS; V tile transposed via packed v2bf stores; V
// fragments loaded once per d-subtile and reused across both q-subtiles.
// ---------------------------------------------------------------------------
__global__ __launch_bounds__(128)
void attn_flash(const bf16_t* __restrict__ Qb, const bf16_t* __restrict__ Kb,
                const bf16_t* __restrict__ Vb, bf16_t* __restrict__ Out) {
    __shared__ __align__(16) bf16_t sK [32 * 72];        // [key][d], 144B stride
    __shared__ __align__(16) bf16_t sVt[64 * 40];        // [d][key], 80B stride
    __shared__ __align__(16) bf16_t sP [4 * 32 * 40];    // per-wave P 32x32

    const int t = threadIdx.x, lane = t & 31, w = t >> 5;
    const int b = blockIdx.z, h = blockIdx.y;
    const int q0 = blockIdx.x * 128 + w * 32;
    const int kvh = h >> 2;                              // n_rep = 4

    const bf16_t* qptr  = Qb + (((size_t)(b * 32 + h)) * 2048 + q0) * 64;
    const bf16_t* kbase = Kb + ((size_t)(b * 8 + kvh)) * 2048 * 64;
    const bf16_t* vbase = Vb + ((size_t)(b * 8 + kvh)) * 2048 * 64;

    v16bf qa[2][2];
    #pragma unroll
    for (int qs = 0; qs < 2; ++qs) {
        qa[qs][0] = load_frag_b16(qptr + qs * 16 * 64, 64);
        qa[qs][1] = load_frag_b16(qptr + qs * 16 * 64 + 32, 64);
    }

    v8f o[2][4];
    float mrow[2][8], lrow[2][8];
    #pragma unroll
    for (int qs = 0; qs < 2; ++qs) {
        #pragma unroll
        for (int j = 0; j < 4; ++j) { v8f z = ZERO8F; o[qs][j] = z; }
        #pragma unroll
        for (int i = 0; i < 8; ++i) { mrow[qs][i] = -3.0e38f; lrow[qs][i] = 0.0f; }
    }
    const float scale = 0.125f;
    const int rb = (lane >> 4) * 8;
    const int cn = lane & 15;

    const int kkey = t >> 2, kc = (t & 3) * 16;          // K: 2x16B per thread
    const int vk = (t >> 3) * 2, vc = (t & 7) * 8;       // V: key pair, 8 d's

    for (int kt = 0; kt < 2048; kt += 32) {
        const bf16_t* gk = kbase + (size_t)(kt + kkey) * 64 + kc;
        async_load_b128(gk,     &sK[kkey * 72 + kc]);
        async_load_b128(gk + 8, &sK[kkey * 72 + kc + 8]);

        v8bf v0 = *(const v8bf*)(vbase + (size_t)(kt + vk) * 64 + vc);
        v8bf v1 = *(const v8bf*)(vbase + (size_t)(kt + vk + 1) * 64 + vc);
        #pragma unroll
        for (int j = 0; j < 8; ++j) {
            v2bf pk; pk[0] = v0[j]; pk[1] = v1[j];
            *(v2bf*)(&sVt[(vc + j) * 40 + vk]) = pk;
        }
        wait_async0();
        __syncthreads();

        // K^T fragments shared across both q-subtiles.
        v16bf kf00 = load_frag_b16(sK, 72);
        v16bf kf01 = load_frag_b16(sK + 32, 72);
        v16bf kf10 = load_frag_b16(sK + 16 * 72, 72);
        v16bf kf11 = load_frag_b16(sK + 16 * 72 + 32, 72);

        #pragma unroll
        for (int qs = 0; qs < 2; ++qs) {
            v8f s0 = ZERO8F, s1 = ZERO8F;
            s0 = wmma_bf16(qa[qs][0], kf00, s0);
            s0 = wmma_bf16(qa[qs][1], kf01, s0);
            s1 = wmma_bf16(qa[qs][0], kf10, s1);
            s1 = wmma_bf16(qa[qs][1], kf11, s1);

            #pragma unroll
            for (int i = 0; i < 8; ++i) {
                float a0 = s0[i] * scale, a1 = s1[i] * scale;
                float mx = fmaxf(a0, a1);
                #pragma unroll
                for (int msk = 1; msk <= 8; msk <<= 1) mx = fmaxf(mx, __shfl_xor(mx, msk, 32));
                float mnew  = fmaxf(mrow[qs][i], mx);
                float alpha = __expf(mrow[qs][i] - mnew);
                float p0 = __expf(a0 - mnew);
                float p1 = __expf(a1 - mnew);
                float rs = p0 + p1;
                #pragma unroll
                for (int msk = 1; msk <= 8; msk <<= 1) rs += __shfl_xor(rs, msk, 32);
                lrow[qs][i] = lrow[qs][i] * alpha + rs;
                mrow[qs][i] = mnew;
                #pragma unroll
                for (int j = 0; j < 4; ++j) o[qs][j][i] *= alpha;
                bf16_t* prow = sP + (w * 32 + qs * 16 + rb + i) * 40;
                prow[cn]      = (bf16_t)p0;
                prow[cn + 16] = (bf16_t)p1;
            }
        }

        // O(32x64) += P(32x32) @ V(32x64); V fragment reused across q-subtiles.
        v16bf pa0 = load_frag_b16(sP + (w * 32) * 40, 40);
        v16bf pa1 = load_frag_b16(sP + (w * 32 + 16) * 40, 40);
        #pragma unroll
        for (int j = 0; j < 4; ++j) {
            v16bf vfj = load_frag_b16(sVt + j * 16 * 40, 40);
            o[0][j] = wmma_bf16(pa0, vfj, o[0][j]);
            o[1][j] = wmma_bf16(pa1, vfj, o[1][j]);
        }
        __syncthreads();
    }

    #pragma unroll
    for (int qs = 0; qs < 2; ++qs)
        #pragma unroll
        for (int i = 0; i < 8; ++i) {
            float inv = 1.0f / lrow[qs][i];
            int srow = q0 + qs * 16 + rb + i;
            bf16_t* orow = Out + ((size_t)b * 2048 + srow) * 2048 + h * 64;
            #pragma unroll
            for (int j = 0; j < 4; ++j) orow[j * 16 + cn] = (bf16_t)(o[qs][j][i] * inv);
        }
}

// ---------------------------------------------------------------------------
extern "C" void kernel_launch(void* const* d_in, const int* in_sizes, int n_in,
                              void* d_out, int out_size, void* d_ws, size_t ws_size,
                              hipStream_t stream) {
    const float* x  = (const float*)d_in[0];
    const float* Wq = (const float*)d_in[1];
    const float* Wk = (const float*)d_in[2];
    const float* Wv = (const float*)d_in[3];
    const float* Wo = (const float*)d_in[4];
    float* out = (float*)d_out;

    char* ws = (char*)d_ws;
    size_t off = 0;
    bf16_t* xb  = (bf16_t*)(ws + off); off += (size_t)4096 * 2048 * 2;   // 16 MB
    bf16_t* Wqb = (bf16_t*)(ws + off); off += (size_t)2048 * 2048 * 2;   //  8 MB
    bf16_t* Wkb = (bf16_t*)(ws + off); off += (size_t)2048 * 512 * 2;    //  2 MB
    bf16_t* Wvb = (bf16_t*)(ws + off); off += (size_t)2048 * 512 * 2;    //  2 MB
    bf16_t* Wob = (bf16_t*)(ws + off); off += (size_t)2048 * 2048 * 2;   //  8 MB
    float*  qf  = (float*)(ws + off);  off += (size_t)4096 * 2048 * 4;   // 32 MB
    float*  kf  = (float*)(ws + off);  off += (size_t)4096 * 512 * 4;    //  8 MB
    float*  vf  = (float*)(ws + off);  off += (size_t)4096 * 512 * 4;    //  8 MB
    bf16_t* Qb  = (bf16_t*)(ws + off); off += (size_t)4096 * 2048 * 2;   // 16 MB
    bf16_t* Kb  = (bf16_t*)(ws + off); off += (size_t)4096 * 512 * 2;    //  4 MB
    bf16_t* Vb  = (bf16_t*)(ws + off); off += (size_t)4096 * 512 * 2;    //  4 MB
    bf16_t* Ab  = (bf16_t*)qf;   // attention bf16 out reuses qf region (dead by then)

    // 1) One-shot bf16 conversion.
    cvt_f32_bf16<<<8192, 256, 0, stream>>>(x,  xb,  4096 * 2048 / 4);
    cvt_f32_bf16<<<4096, 256, 0, stream>>>(Wq, Wqb, 2048 * 2048 / 4);
    cvt_f32_bf16<<<1024, 256, 0, stream>>>(Wk, Wkb, 2048 * 512 / 4);
    cvt_f32_bf16<<<1024, 256, 0, stream>>>(Wv, Wvb, 2048 * 512 / 4);
    cvt_f32_bf16<<<4096, 256, 0, stream>>>(Wo, Wob, 2048 * 2048 / 4);

    // 2) QKV projections (WMMA), M-tile 256, N-tile 128.
    dim3 blk(256);
    gemm_bf16_wmma<<<dim3(16, 16), blk, 0, stream>>>(xb, Wqb, qf, 4096, 2048, 2048);
    gemm_bf16_wmma<<<dim3( 4, 16), blk, 0, stream>>>(xb, Wkb, kf, 4096,  512, 2048);
    gemm_bf16_wmma<<<dim3( 4, 16), blk, 0, stream>>>(xb, Wvb, vf, 4096,  512, 2048);

    // 3) RoPE + pack.
    rope_pack<<<4096, 128, 0, stream>>>(qf, kf, vf, Qb, Kb, Vb);

    // 4) Attention (WMMA flash, writes bf16).
    attn_flash<<<dim3(16, 32, 2), 128, 0, stream>>>(Qb, Kb, Vb, Ab);

    // 5) Output projection (WMMA).
    gemm_bf16_wmma<<<dim3(16, 16), blk, 0, stream>>>(Ab, Wob, out, 4096, 2048, 2048);
}
